// GCN_343597384437
// MI455X (gfx1250) — matrix operand order repaired and checked by reading, not compiled, not run on hardware
//
#include <hip/hip_runtime.h>
#include <math.h>

typedef __attribute__((ext_vector_type(16))) _Float16 v16h;
typedef __attribute__((ext_vector_type(8)))  float    v8f;

// ---------------------------------------------------------------------------
// Degree / normalization
// ---------------------------------------------------------------------------
__global__ void k_init_deg(float* __restrict__ deg, int n) {
  int i = blockIdx.x * blockDim.x + threadIdx.x;
  if (i < n) deg[i] = 1.0f;  // self-loop contributes 1
}

__global__ void k_deg_edges(const long long* __restrict__ ei, long long E,
                            float* __restrict__ deg) {
  long long e = (long long)blockIdx.x * blockDim.x + threadIdx.x;
  if (e < E) atomicAdd(&deg[(int)ei[E + e]], 1.0f);  // dst = row 1
}

__global__ void k_dinv(const float* __restrict__ deg, float* __restrict__ dinv, int n) {
  int i = blockIdx.x * blockDim.x + threadIdx.x;
  if (i < n) { float d = deg[i]; dinv[i] = d > 0.f ? rsqrtf(d) : 0.f; }
}

// ---------------------------------------------------------------------------
// WMMA GEMM: out[n x FOUT] = act(A) @ W, act = (BN scale/shift + ReLU) if
// scale != nullptr. A fp32 -> f16 at fragment build; f32 accumulate via
// v_wmma_f32_16x16x32_f16. One wave per 16-row M tile.
//
// W is staged in LDS PRE-SWIZZLED into B-fragment order:
//   sWf[(((nt*KT + kt)*32 + lane)*16) + e]  with  k = kt*32 + (lane>>4)*16 + e,
//                                                 n = nt*16 + (lane&15)
// so each wave's B fragment is a single contiguous 32B per-lane LDS load
// (2x ds_load_b128) instead of 16 scalar ds_load_u16 + b16 packing.
// ---------------------------------------------------------------------------
template <int FIN, int FOUT, int FOUT_PAD>
__global__ __launch_bounds__(256) void k_gemm_wmma(
    const float* __restrict__ A, const float* __restrict__ W,
    const float* __restrict__ scale, const float* __restrict__ shift,
    float* __restrict__ out, int nrows) {
  constexpr int KT = FIN / 32;        // K tiles of 32
  constexpr int NT = FOUT_PAD / 16;   // N tiles of 16
  __shared__ _Float16 sWf[NT * KT * 32 * 16];  // == FIN * FOUT_PAD halfs

  // Fill LDS in fragment order. Contiguous LDS writes per thread; W reads are
  // scattered but W is tiny (<=64KB) and L2-resident.
  for (int i = threadIdx.x; i < NT * KT * 32 * 16; i += 256) {
    int e    = i & 15;
    int ln   = (i >> 4) & 31;
    int t    = i >> 9;
    int kt   = t % KT;
    int nt   = t / KT;
    int k    = kt * 32 + ((ln >> 4) << 4) + e;  // contiguous k order (same as A): lane half = K half
    int nn   = nt * 16 + (ln & 15);
    sWf[i] = (nn < FOUT) ? (_Float16)W[k * FOUT + nn] : (_Float16)0.f;
  }
  __syncthreads();

  const int wave = threadIdx.x >> 5;
  const int lane = threadIdx.x & 31;
  const int rowTile = (blockIdx.x * 8 + wave) * 16;
  if (rowTile >= nrows) return;  // wave-uniform: EXEC stays all-ones for WMMA

  const int m  = lane & 15;   // A row within tile / C column
  const int kg = lane >> 4;   // lane half
  const float* __restrict__ arow = A + (size_t)(rowTile + m) * FIN;

  // A fragments for all K tiles, fused BN+ReLU, loaded as float2 pairs
  // (adjacent e pairs are adjacent K values in the ISA A layout).
  v16h afrag[KT];
  for (int kt = 0; kt < KT; ++kt) {
#pragma unroll
    for (int v = 0; v < 8; ++v) {
      int kk = kt * 32 + (kg << 4) + (v << 1);
      float2 xy = *reinterpret_cast<const float2*>(arow + kk);
      if (scale) {
        xy.x = xy.x * scale[kk] + shift[kk];
        xy.y = xy.y * scale[kk + 1] + shift[kk + 1];
        xy.x = xy.x > 0.f ? xy.x : 0.f;
        xy.y = xy.y > 0.f ? xy.y : 0.f;
      }
      afrag[kt][2 * v]     = (_Float16)xy.x;
      afrag[kt][2 * v + 1] = (_Float16)xy.y;
    }
  }

  const v16h* __restrict__ bfrag =
      reinterpret_cast<const v16h*>(sWf) + lane;  // + (nt*KT+kt)*32
  for (int nt = 0; nt < NT; ++nt) {
    v8f c = {};
#pragma unroll
    for (int kt = 0; kt < KT; ++kt) {
      v16h b = bfrag[(nt * KT + kt) * 32];  // one 32B contiguous LDS load
      c = __builtin_amdgcn_wmma_f32_16x16x32_f16(
          /*neg_a=*/false, afrag[kt], /*neg_b=*/false, b,
          /*c_mod=*/(short)0, c, /*reuse_a=*/false, /*reuse_b=*/false);
    }
#pragma unroll
    for (int r = 0; r < 8; ++r) {  // C/D: VGPR r holds row r + 8*lane_half
      int row = rowTile + r + (kg << 3);
      int col = nt * 16 + m;
      if (col < FOUT) out[(size_t)row * FOUT + col] = c[r];
    }
  }
}

// ---------------------------------------------------------------------------
// agg init: bias + self-loop term (norm_ii = dinv[i]^2)
// ---------------------------------------------------------------------------
template <int F>
__global__ void k_init_agg(const float* __restrict__ xw, const float* __restrict__ dinv,
                           const float* __restrict__ bias, float* __restrict__ agg,
                           int n) {
  long long idx = (long long)blockIdx.x * blockDim.x + threadIdx.x;
  if (idx >= (long long)n * F) return;
  int i = (int)(idx / F), f = (int)(idx % F);
  float w = dinv[i];
  agg[idx] = bias[f] + w * w * xw[idx];
}

// ---------------------------------------------------------------------------
// Edge scatter: agg[dst] += dinv[src]*dinv[dst] * xw[src]; float4 per lane,
// one wave covers one edge when F==128 (coalesced row read + atomics, L2-hot).
// ---------------------------------------------------------------------------
template <int F>
__global__ void k_scatter(const float* __restrict__ xw, const float* __restrict__ dinv,
                          const long long* __restrict__ ei, long long E,
                          float* __restrict__ agg) {
  constexpr int CPE = F / 4;
  long long gid = (long long)blockIdx.x * blockDim.x + threadIdx.x;
  if (gid >= E * CPE) return;
  long long e = gid / CPE;
  int c = (int)(gid % CPE);
  int s = (int)ei[e], d = (int)ei[E + e];
  float w = dinv[s] * dinv[d];
  float4 v = *reinterpret_cast<const float4*>(xw + (size_t)s * F + 4 * c);
  float* o = agg + (size_t)d * F + 4 * c;
  atomicAdd(o + 0, w * v.x);
  atomicAdd(o + 1, w * v.y);
  atomicAdd(o + 2, w * v.z);
  atomicAdd(o + 3, w * v.w);
}

// ---------------------------------------------------------------------------
// BatchNorm statistics (sum / sumsq per column, 128 columns), then fold into
// per-feature scale/shift so the next GEMM applies BN+ReLU for free.
// ---------------------------------------------------------------------------
__global__ void k_zero(float* __restrict__ p, int n) {
  int i = blockIdx.x * blockDim.x + threadIdx.x;
  if (i < n) p[i] = 0.f;
}

__global__ void k_bn_partial(const float* __restrict__ h, int n,
                             float* __restrict__ stats) {
  int tid = blockIdx.x * blockDim.x + threadIdx.x;
  int c = tid & 127;
  int r0 = tid >> 7;
  int stride = (gridDim.x * blockDim.x) >> 7;
  float s = 0.f, s2 = 0.f;
  for (int r = r0; r < n; r += stride) {
    float v = h[(size_t)r * 128 + c];  // consecutive tids -> consecutive c: coalesced
    s += v; s2 += v * v;
  }
  atomicAdd(&stats[c], s);
  atomicAdd(&stats[128 + c], s2);
}

__global__ void k_bn_finalize(const float* __restrict__ stats,
                              const float* __restrict__ g, const float* __restrict__ be,
                              float n, float* __restrict__ scale,
                              float* __restrict__ shift) {
  int c = threadIdx.x;  // 128 threads
  float mean = stats[c] / n;
  float var  = stats[128 + c] / n - mean * mean;
  float s = g[c] * rsqrtf(var + 1e-5f);
  scale[c] = s;
  shift[c] = be[c] - mean * s;
}

// ---------------------------------------------------------------------------
// log_softmax over 40 classes, one wave32 per node.
// ---------------------------------------------------------------------------
__global__ void k_logsoftmax40(const float* __restrict__ h, float* __restrict__ out,
                               int n) {
  int gid = blockIdx.x * blockDim.x + threadIdx.x;
  int node = gid >> 5, lane = gid & 31;
  if (node >= n) return;
  const float* row = h + (size_t)node * 40;
  float a = row[lane];
  float b = (lane < 8) ? row[lane + 32] : -3.0e38f;
  float m = fmaxf(a, b);
  for (int off = 16; off; off >>= 1) m = fmaxf(m, __shfl_xor(m, off, 32));
  float s = expf(a - m) + ((lane < 8) ? expf(b - m) : 0.f);
  for (int off = 16; off; off >>= 1) s += __shfl_xor(s, off, 32);
  float l = m + logf(s);
  out[(size_t)node * 40 + lane] = a - l;
  if (lane < 8) out[(size_t)node * 40 + lane + 32] = b - l;
}

// ---------------------------------------------------------------------------
extern "C" void kernel_launch(void* const* d_in, const int* in_sizes, int n_in,
                              void* d_out, int out_size, void* d_ws, size_t ws_size,
                              hipStream_t stream) {
  (void)n_in; (void)out_size; (void)ws_size;
  const float*     x   = (const float*)d_in[0];
  const long long* ei  = (const long long*)d_in[1];
  const float*     W0  = (const float*)d_in[2];
  const float*     b0  = (const float*)d_in[3];
  const float*     g0  = (const float*)d_in[4];
  const float*     be0 = (const float*)d_in[5];
  const float*     W1  = (const float*)d_in[6];
  const float*     b1  = (const float*)d_in[7];
  const float*     g1  = (const float*)d_in[8];
  const float*     be1 = (const float*)d_in[9];
  const float*     W2  = (const float*)d_in[10];
  const float*     b2  = (const float*)d_in[11];

  const int n = in_sizes[0] / 128;
  const long long E = in_sizes[1] / 2;

  // Bump-allocate workspace (256B aligned).
  char* p = (char*)d_ws;
  auto alloc = [&](size_t bytes) -> void* {
    char* r = p;
    p += (bytes + 255) & ~(size_t)255;
    return r;
  };
  float* deg   = (float*)alloc((size_t)n * 4);
  float* dinv  = (float*)alloc((size_t)n * 4);
  float* XW    = (float*)alloc((size_t)n * 128 * 4);  // reused as N x 40 in layer 2
  float* AGG   = (float*)alloc((size_t)n * 128 * 4);  // reused as N x 40 in layer 2
  float* stats = (float*)alloc(256 * 4);
  float* sc    = (float*)alloc(128 * 4);
  float* sh    = (float*)alloc(128 * 4);

  const dim3 B(256);
  const int gN    = (n + 255) / 256;
  const int gEdge = (int)((E + 255) / 256);
  const int gGemm = (n + 127) / 128;
  const int gNF   = (int)(((long long)n * 128 + 255) / 256);
  const int gN40  = (int)(((long long)n * 40 + 255) / 256);
  const int gS128 = (int)((E * 32 + 255) / 256);
  const int gS40  = (int)((E * 10 + 255) / 256);
  const int gLSM  = (int)(((long long)n * 32 + 255) / 256);

  // Normalization: deg (with self-loop), dinv = rsqrt(deg)
  k_init_deg<<<gN, B, 0, stream>>>(deg, n);
  k_deg_edges<<<gEdge, B, 0, stream>>>(ei, E, deg);
  k_dinv<<<gN, B, 0, stream>>>(deg, dinv, n);

  // Layer 0: xw0 = x @ W0 ; agg0 = scatter + b0
  k_gemm_wmma<128, 128, 128><<<gGemm, B, 0, stream>>>(x, W0, nullptr, nullptr, XW, n);
  k_init_agg<128><<<gNF, B, 0, stream>>>(XW, dinv, b0, AGG, n);
  k_scatter<128><<<gS128, B, 0, stream>>>(XW, dinv, ei, E, AGG);
  k_zero<<<1, B, 0, stream>>>(stats, 256);
  k_bn_partial<<<512, B, 0, stream>>>(AGG, n, stats);
  k_bn_finalize<<<1, 128, 0, stream>>>(stats, g0, be0, (float)n, sc, sh);

  // Layer 1: BN+ReLU fused into GEMM A-load
  k_gemm_wmma<128, 128, 128><<<gGemm, B, 0, stream>>>(AGG, W1, sc, sh, XW, n);
  k_init_agg<128><<<gNF, B, 0, stream>>>(XW, dinv, b1, AGG, n);
  k_scatter<128><<<gS128, B, 0, stream>>>(XW, dinv, ei, E, AGG);
  k_zero<<<1, B, 0, stream>>>(stats, 256);
  k_bn_partial<<<512, B, 0, stream>>>(AGG, n, stats);
  k_bn_finalize<<<1, 128, 0, stream>>>(stats, g1, be1, (float)n, sc, sh);

  // Layer 2: FOUT = 40 (padded to 48 inside the kernel), then log_softmax
  k_gemm_wmma<128, 40, 48><<<gGemm, B, 0, stream>>>(AGG, W2, sc, sh, XW, n);
  k_init_agg<40><<<gN40, B, 0, stream>>>(XW, dinv, b2, AGG, n);
  k_scatter<40><<<gS40, B, 0, stream>>>(XW, dinv, ei, E, AGG);
  k_logsoftmax40<<<gLSM, B, 0, stream>>>(AGG, (float*)d_out, n);
}
